// GAT_MS_2_3_33251636806234
// MI455X (gfx1250) — matrix operand order repaired and checked by reading, not compiled, not run on hardware
//
#include <hip/hip_runtime.h>
#include <hip/hip_bf16.h>

// MI455X / gfx1250 implementation of GAT_MS_2_3 mean-shift attention.
// Strategy: bf16 WMMA (v_wmma_f32_16x16x32_bf16) for all GEMMs; fused
// flash-style mean-shift iteration with one wave32 per head; the softmax
// denominator and the xyz mean-shift are computed by a single extra WMMA
// whose B operand packs [ones | x | y | z]. All big operands fit in the
// 192MB L2, so per-head re-reads of delta_y/masks are L2 hits.

typedef __attribute__((ext_vector_type(16))) __bf16 v16bf;
typedef __attribute__((ext_vector_type(8)))  __bf16 v8bf;
typedef __attribute__((ext_vector_type(8)))  float  v8f;

#define N_TOK 3072
#define DIM   256
#define HEADS 8
#define DH    32
#define WMAT_ELEMS (DIM*DIM)

static __device__ __forceinline__ v16bf ld16(const __bf16* p0, const __bf16* p1) {
    v16bf r;
    *(v8bf*)&r       = *(const v8bf*)p0;
    *((v8bf*)&r + 1) = *(const v8bf*)p1;
    return r;
}

static __device__ __forceinline__ v8f wmma_bf16(v16bf a, v16bf b, v8f c) {
    // (neg_a, A, neg_b, B, c_mod, C, reuse_a, reuse_b)
    return __builtin_amdgcn_wmma_f32_16x16x32_bf16(false, a, false, b, (short)0, c, false, false);
}

// ---------------------------------------------------------------------------
// Convert x -> bf16 and the four weight matrices -> transposed bf16 [out][in]
// (B operands of WMMA want K-contiguous, i.e. [N][K] storage).
// ---------------------------------------------------------------------------
__global__ void k_cvt(const float* __restrict__ x,
                      const float* __restrict__ Wq, const float* __restrict__ Wk,
                      const float* __restrict__ Wv, const float* __restrict__ Wo,
                      __bf16* __restrict__ xb, __bf16* __restrict__ WT) {
    int idx = blockIdx.x * 256 + threadIdx.x;
    if (idx < N_TOK * DIM) {
        xb[idx] = (__bf16)x[idx];
    } else {
        int e  = idx - N_TOK * DIM;           // 4 * 65536 weight elements
        int wi = e >> 16;
        int r  = (e & 65535) >> 8;            // output feature
        int c  = e & 255;                     // input feature
        const float* W = (wi == 0) ? Wq : (wi == 1) ? Wk : (wi == 2) ? Wv : Wo;
        WT[wi * WMAT_ELEMS + r * DIM + c] = (__bf16)W[c * DIM + r];  // transpose
    }
}

// ---------------------------------------------------------------------------
// QKV projection: 8 waves/block, each wave one 16x16 output tile, K-loop of 8
// WMMAs.  q,k stored row-major bf16 [tok][D]; v stored transposed [D][tok]
// (contiguous v8bf store straight out of the WMMA C layout).
// ---------------------------------------------------------------------------
__global__ void __launch_bounds__(256) k_qkv(const __bf16* __restrict__ xb,
                                             const __bf16* __restrict__ WT,
                                             const float* __restrict__ bq,
                                             const float* __restrict__ bk,
                                             const float* __restrict__ bv,
                                             __bf16* __restrict__ qb,
                                             __bf16* __restrict__ kb,
                                             __bf16* __restrict__ vT) {
    int lane = threadIdx.x & 31, w = threadIdx.x >> 5;
    int hi = lane >> 4, ln = lane & 15;
    int r0 = blockIdx.x * 16;
    int c0 = (blockIdx.y * 8 + w) * 16;
    int z  = blockIdx.z;

    const __bf16* Wt   = WT + z * WMAT_ELEMS;
    const float*  bias = (z == 0) ? bq : (z == 1) ? bk : bv;

    const __bf16* arow = xb + (r0 + ln) * DIM + hi * 16;   // A: [M][K] row-major
    const __bf16* brow = Wt + (c0 + ln) * DIM + hi * 16;   // B: [N][K] row-major

    v8f acc = {};
    for (int kk = 0; kk < DIM; kk += 32) {
        v16bf A = ld16(arow + kk, arow + kk + 8);
        v16bf B = ld16(brow + kk, brow + kk + 8);
        acc = wmma_bf16(A, B, acc);
    }
    float b = bias[c0 + ln];
    if (z < 2) {
        __bf16* out = (z == 0) ? qb : kb;
        #pragma unroll
        for (int g = 0; g < 8; ++g)
            out[(r0 + g + 8 * hi) * DIM + c0 + ln] = (__bf16)(acc[g] + b);
    } else {
        v8bf o;
        #pragma unroll
        for (int g = 0; g < 8; ++g) o[g] = (__bf16)(acc[g] + b);
        *(v8bf*)(vT + (c0 + ln) * N_TOK + r0 + 8 * hi) = o;  // transposed store
    }
}

// ---------------------------------------------------------------------------
// Per-iteration prep: build xyzT (row0=1, rows1-3 = coords, rows4-15 = 0) for
// the combined rowsum+xyz WMMA B operand, and zero the xyz accumulator.
// ---------------------------------------------------------------------------
__global__ void k_prep(const float* __restrict__ cur,
                       float* __restrict__ next,
                       __bf16* __restrict__ xyzT) {
    int n = blockIdx.x * 256 + threadIdx.x;
    if (n >= N_TOK) return;
    xyzT[n] = (__bf16)1.0f;
    #pragma unroll
    for (int c = 0; c < 3; ++c) {
        xyzT[(1 + c) * N_TOK + n] = (__bf16)cur[n * 3 + c];
        next[n * 3 + c] = 0.0f;
    }
    #pragma unroll
    for (int r = 4; r < 16; ++r) xyzT[r * N_TOK + n] = (__bf16)0.0f;
}

// ---------------------------------------------------------------------------
// One mean-shift iteration.  Block = 16 query rows, wave w == head w.
// Per 32-key step: 2 score WMMAs -> gated exp -> P via LDS relayout ->
// 2 P@V WMMAs + 1 combined (rowsum|xyz) WMMA.  5 WMMAs / 32 keys / wave.
// ---------------------------------------------------------------------------
__global__ void __launch_bounds__(256) k_ms(const __bf16* __restrict__ qb,
                                            const __bf16* __restrict__ kb,
                                            const __bf16* __restrict__ vT,
                                            const __bf16* __restrict__ xyzT,
                                            const float* __restrict__ xyz,
                                            const float* __restrict__ dy,
                                            const unsigned char* __restrict__ dmask,
                                            const unsigned char* __restrict__ bmask,
                                            __bf16* __restrict__ featb,
                                            float* __restrict__ xyz_next) {
    __shared__ __bf16 Pl[HEADS][16 * 32];   // per-wave P staging (C->A relayout)

    int lane = threadIdx.x & 31, h = threadIdx.x >> 5;
    int hi = lane >> 4, ln = lane & 15;
    int r0 = blockIdx.x * 16;

    // A operand: q tile 16x32 (row-major, K=Dh contiguous)
    const __bf16* qrow = qb + (r0 + ln) * DIM + h * DH + hi * 16;
    v16bf Aq = ld16(qrow, qrow + 8);

    // per-lane query coordinates for the 8 C rows it owns
    float q3[8][3];
    #pragma unroll
    for (int g = 0; g < 8; ++g)
        #pragma unroll
        for (int c = 0; c < 3; ++c)
            q3[g][c] = xyz[(r0 + g + 8 * hi) * 3 + c];

    const float scl = 0.17677669529663687f;  // T / sqrt(Dh)
    v8f O0 = {}, O1 = {}, CE = {};
    __bf16* myP = &Pl[h][0];

    for (int m0 = 0; m0 < N_TOK; m0 += 32) {
        // --- scores + gated exp for two 16-key tiles ---
        #pragma unroll
        for (int t = 0; t < 2; ++t) {
            const __bf16* krow = kb + (m0 + t * 16 + ln) * DIM + h * DH + hi * 16;
            v16bf Bk = ld16(krow, krow + 8);
            v8f zero = {};
            v8f s = wmma_bf16(Aq, Bk, zero);
            int key = m0 + t * 16 + ln;
            float kx = xyz[key * 3 + 0], ky = xyz[key * 3 + 1], kz = xyz[key * 3 + 2];
            #pragma unroll
            for (int g = 0; g < 8; ++g) {
                int qr = r0 + g + 8 * hi;
                float dx = q3[g][0] - kx, dv = q3[g][1] - ky, dz = q3[g][2] - kz;
                float d2 = (qr == key) ? 1.0e-4f : (dx * dx + dv * dv + dz * dz);
                long  pi = (long)qr * N_TOK + key;
                bool  valid = ((dmask[pi] & bmask[pi]) != 0) || (qr == key);
                float lg = valid ? (scl * s[g] - 0.25f * d2 - dy[pi]) : -1.0e9f;
                myP[(g + 8 * hi) * 32 + t * 16 + ln] = (__bf16)__expf(lg);
            }
        }
        // --- reload P in A layout (same wave; LDS ops are in-order per wave) ---
        const __bf16* prow = myP + ln * 32 + hi * 16;
        v16bf Ap = ld16(prow, prow + 8);
        // --- O += P @ V  (two Dh column tiles, vT is [D][N] K-contiguous) ---
        const __bf16* v0 = vT + (h * DH + ln) * N_TOK + m0 + hi * 16;
        O0 = wmma_bf16(Ap, ld16(v0, v0 + 8), O0);
        const __bf16* v1 = vT + (h * DH + 16 + ln) * N_TOK + m0 + hi * 16;
        O1 = wmma_bf16(Ap, ld16(v1, v1 + 8), O1);
        // --- rowsum (col0) and xyz accumulation (cols1-3) in one WMMA ---
        const __bf16* e0 = xyzT + ln * N_TOK + m0 + hi * 16;
        CE = wmma_bf16(Ap, ld16(e0, e0 + 8), CE);
    }

    // softmax denominator broadcast: rowsum for row g+8*hi lives in lane hi*16
    float inv[8];
    #pragma unroll
    for (int g = 0; g < 8; ++g) {
        float rs = __shfl(CE[g], hi * 16, 32);
        inv[g] = 1.0f / rs;
    }
    #pragma unroll
    for (int g = 0; g < 8; ++g) {
        int qr = r0 + g + 8 * hi;
        featb[qr * DIM + h * DH + ln]      = (__bf16)(O0[g] * inv[g]);
        featb[qr * DIM + h * DH + 16 + ln] = (__bf16)(O1[g] * inv[g]);
    }
    if (ln >= 1 && ln <= 3) {            // lanes holding x/y/z columns
        #pragma unroll
        for (int g = 0; g < 8; ++g) {
            int qr = r0 + g + 8 * hi;
            atomicAdd(&xyz_next[qr * 3 + (ln - 1)], CE[g] * inv[g] * (1.0f / HEADS));
        }
    }
}

// ---------------------------------------------------------------------------
// Final projection: out = feat @ Wo + bo + x (f32 output with residual).
// ---------------------------------------------------------------------------
__global__ void __launch_bounds__(256) k_out(const __bf16* __restrict__ featb,
                                             const __bf16* __restrict__ WT,
                                             const float* __restrict__ bo,
                                             const float* __restrict__ x,
                                             float* __restrict__ out) {
    int lane = threadIdx.x & 31, w = threadIdx.x >> 5;
    int hi = lane >> 4, ln = lane & 15;
    int r0 = blockIdx.x * 16;
    int c0 = (blockIdx.y * 8 + w) * 16;
    const __bf16* Wt = WT + 3 * WMAT_ELEMS;

    const __bf16* arow = featb + (r0 + ln) * DIM + hi * 16;
    const __bf16* brow = Wt + (c0 + ln) * DIM + hi * 16;

    v8f acc = {};
    for (int kk = 0; kk < DIM; kk += 32) {
        v16bf A = ld16(arow + kk, arow + kk + 8);
        v16bf B = ld16(brow + kk, brow + kk + 8);
        acc = wmma_bf16(A, B, acc);
    }
    float b = bo[c0 + ln];
    #pragma unroll
    for (int g = 0; g < 8; ++g) {
        int qr = r0 + g + 8 * hi;
        out[qr * DIM + c0 + ln] = acc[g] + b + x[qr * DIM + c0 + ln];
    }
}

// ---------------------------------------------------------------------------
// Host side.  Inputs (setup_inputs order):
//  0 x[N,D] f32, 1 xyz[N,3] f32, 2 delta_y[N,N] f32, 3 distance_mask[N,N] bool,
//  4 big_inter_mask[N,N] bool, 5 Wq, 6 bq, 7 Wk, 8 bk, 9 Wv, 10 bv, 11 Wo, 12 bo.
// Output: xyz_out (N*3 f32) then out (N*D f32), concatenated.
// Workspace use ~8.6 MB.
// ---------------------------------------------------------------------------
extern "C" void kernel_launch(void* const* d_in, const int* in_sizes, int n_in,
                              void* d_out, int out_size, void* d_ws, size_t ws_size,
                              hipStream_t stream) {
    const float* x    = (const float*)d_in[0];
    const float* xyz  = (const float*)d_in[1];
    const float* dy   = (const float*)d_in[2];
    const unsigned char* dm = (const unsigned char*)d_in[3];
    const unsigned char* bm = (const unsigned char*)d_in[4];
    const float* Wq = (const float*)d_in[5];  const float* bq = (const float*)d_in[6];
    const float* Wk = (const float*)d_in[7];  const float* bk = (const float*)d_in[8];
    const float* Wv = (const float*)d_in[9];  const float* bv = (const float*)d_in[10];
    const float* Wo = (const float*)d_in[11]; const float* bo = (const float*)d_in[12];

    char* ws = (char*)d_ws;
    size_t off = 0;
    __bf16* xb    = (__bf16*)(ws + off); off += (size_t)N_TOK * DIM * 2;   // 1.5 MB
    __bf16* WT    = (__bf16*)(ws + off); off += (size_t)4 * WMAT_ELEMS * 2;// 0.5 MB
    __bf16* qb    = (__bf16*)(ws + off); off += (size_t)N_TOK * DIM * 2;
    __bf16* kb    = (__bf16*)(ws + off); off += (size_t)N_TOK * DIM * 2;
    __bf16* vT    = (__bf16*)(ws + off); off += (size_t)N_TOK * DIM * 2;
    __bf16* featb = (__bf16*)(ws + off); off += (size_t)N_TOK * DIM * 2;
    float*  xyzA  = (float*)(ws + off);  off += (size_t)N_TOK * 3 * 4;
    float*  xyzB  = (float*)(ws + off);  off += (size_t)N_TOK * 3 * 4;
    __bf16* xyzT  = (__bf16*)(ws + off); off += (size_t)16 * N_TOK * 2;

    float* out_xyz  = (float*)d_out;                 // [N,3]
    float* out_feat = out_xyz + N_TOK * 3;           // [N,D]

    // 1. convert x + transposed weights to bf16
    k_cvt<<<dim3((N_TOK * DIM + 4 * WMAT_ELEMS) / 256), dim3(256), 0, stream>>>(
        x, Wq, Wk, Wv, Wo, xb, WT);

    // 2. q,k,v projections
    k_qkv<<<dim3(N_TOK / 16, 2, 3), dim3(256), 0, stream>>>(
        xb, WT, bq, bk, bv, qb, kb, vT);

    // 3. three mean-shift iterations (xyz ping-pong; last writes d_out)
    const float* cur[3] = { xyz, xyzA, xyzB };
    float*       nxt[3] = { xyzA, xyzB, out_xyz };
    for (int it = 0; it < 3; ++it) {
        k_prep<<<dim3(N_TOK / 256), dim3(256), 0, stream>>>(cur[it], nxt[it], xyzT);
        k_ms<<<dim3(N_TOK / 16), dim3(256), 0, stream>>>(
            qb, kb, vT, xyzT, cur[it], dy, dm, bm, featb, nxt[it]);
    }

    // 4. output projection + residual
    k_out<<<dim3(N_TOK / 16, 2), dim3(256), 0, stream>>>(featb, WT, bo, x, out_feat);
}
